// ResidualGCN_64287070486858
// MI455X (gfx1250) — compile-verified, matrix operand order checked by reading, not run on hardware
//
#include <hip/hip_runtime.h>
#include <hip/hip_bf16.h>
#include <stdint.h>

// ---------------------------------------------------------------------------
// Residual GCN for MI455X (gfx1250).
//  * Dense GEMMs on v_wmma_f32_16x16x32_bf16 (bf16 in, fp32 accumulate).
//  * GEMM tile staging via global_load_async_to_lds_b128 (CDNA5 async DMA,
//    ASYNCcnt-tracked, no VGPR round-trip), s_wait_asynccnt + barrier.
//  * bf16 mirrors of activations so staging is pure b128 DMA.
//  * LDS tile rows padded to 80B / 1040B for conflict-free ds_load_b128.
//  * Scatter = one wave32 per (edge|self-loop), b128 gathers + fp32 atomics.
// ---------------------------------------------------------------------------

#define N_NODES 132000
#define N_EDGES 1056000
#define N_GRAPH 132

typedef __attribute__((ext_vector_type(16))) __bf16 v16bf;
typedef __attribute__((ext_vector_type(8)))  float  v8f;

union FragAB { v16bf v; unsigned u[8]; };

__device__ __forceinline__ unsigned short f2bf(float x) {
  unsigned u = __float_as_uint(x);
  u += 0x7FFFu + ((u >> 16) & 1u);           // round-to-nearest-even
  return (unsigned short)(u >> 16);
}
__device__ __forceinline__ float lrelu(float v) { return v > 0.f ? v : 0.01f * v; }

// Async 16B copy global->LDS (CDNA5 TDM-lite path, tracked by ASYNCcnt).
// LDS destination = low 32 bits of the generic shared-memory address
// (flat->LDS mapping, ISA 10.2).
__device__ __forceinline__ void async_b128(const unsigned short* lds_dst,
                                           const unsigned short* gsrc) {
  unsigned ldsoff = (unsigned)(uintptr_t)lds_dst;
  asm volatile("global_load_async_to_lds_b128 %0, %1, off"
               :: "v"(ldsoff), "v"(gsrc)
               : "memory");
}
__device__ __forceinline__ void wait_async() {
  asm volatile("s_wait_asynccnt 0x0" ::: "memory");
}

// 16x32 bf16 WMMA A/B fragment from an LDS row (CDNA5 layout):
// lane L holds row/col (L&15); VGPR v holds K pair (v>>2)*16 + (L>>4)*8 + (v&3)*2.
// Row base must be 16B aligned; reads dwords [o..o+3] and [8+o..8+o+3].
__device__ __forceinline__ v16bf ld_frag(const unsigned short* row, int half) {
  FragAB f;
  const unsigned* p = reinterpret_cast<const unsigned*>(row);
  int o = half * 4;
#pragma unroll
  for (int i = 0; i < 4; ++i) { f.u[i] = p[o + i]; f.u[4 + i] = p[8 + o + i]; }
  return f.v;
}

__device__ __forceinline__ v8f v8f_zero() {
  v8f z = {0.f, 0.f, 0.f, 0.f, 0.f, 0.f, 0.f, 0.f};
  return z;
}

// --------------------------- weight / input conversion ---------------------
// w: fp32 [K][Ncol] row-major  ->  wt: bf16 [Ncol][Kpad], zero-padded K tail.
__global__ void convert_wt_kernel(const float* __restrict__ w,
                                  unsigned short* __restrict__ wt,
                                  int K, int Kpad, int Ncol) {
  int i = blockIdx.x * blockDim.x + threadIdx.x;
  int total = Ncol * Kpad;
  if (i >= total) return;
  int n = i / Kpad, k = i - n * Kpad;
  wt[i] = (k < K) ? f2bf(w[(size_t)k * Ncol + n]) : (unsigned short)0;
}

// x fp32 [N][140] -> xb bf16 [N][160] (zero-padded)
__global__ void convert_x_kernel(const float* __restrict__ x,
                                 unsigned short* __restrict__ xb) {
  int i = blockIdx.x * blockDim.x + threadIdx.x;
  if (i >= N_NODES * 160) return;
  int n = i / 160, k = i - n * 160;
  xb[i] = (k < 140) ? f2bf(x[(size_t)n * 140 + k]) : (unsigned short)0;
}

// ----------------------------- degree / norm -------------------------------
__global__ void deg_init_kernel(float* __restrict__ deg) {
  int i = blockIdx.x * blockDim.x + threadIdx.x;
  if (i < N_NODES) deg[i] = 1.0f;                     // self loop
}
__global__ void deg_edge_kernel(const int* __restrict__ dst, float* __restrict__ deg) {
  int e = blockIdx.x * blockDim.x + threadIdx.x;
  if (e < N_EDGES) atomicAdd(&deg[dst[e]], 1.0f);
}
__global__ void deg_fin_kernel(float* __restrict__ deg) {
  int i = blockIdx.x * blockDim.x + threadIdx.x;
  if (i < N_NODES) deg[i] = rsqrtf(deg[i]);           // deg >= 1 always
}

// ------------------------------ fused prenet -------------------------------
// h = lrelu( lrelu(x @ W1 + b1) @ W2 + b2 ), 16 node rows per block, 8 waves.
// LDS tile rows padded: KTP=40 elems (80B), intermediate PHP=520 elems (1040B).
__global__ __launch_bounds__(256)
void prenet_kernel(const unsigned short* __restrict__ xb,   // [N][160] bf16
                   const unsigned short* __restrict__ w1t,  // [512][160] bf16
                   const float* __restrict__ pb1,
                   const unsigned short* __restrict__ w2t,  // [256][512] bf16
                   const float* __restrict__ pb2,
                   float* __restrict__ h,
                   unsigned short* __restrict__ hb) {
  constexpr int KP1 = 160, PH_ = 512, PO_ = 256, KT = 32, KTP = 40, PHP = 520;
  __shared__ unsigned short lds_a[16 * KTP];     // 1.25 KB
  __shared__ unsigned short lds_w[PH_ * KTP];    // 40 KB (reused for W2 slices)
  __shared__ unsigned short lds_t[16 * PHP];     // 16.25 KB intermediate tile

  const int tid = threadIdx.x, lane = tid & 31, wave = tid >> 5;
  const int m15 = lane & 15, half = lane >> 4;
  const int row0 = blockIdx.x * 16;

  // ---- stage A: t = lrelu(x @ W1 + b1), wave w owns cols [64w, 64w+64) ----
  v8f c1[4];
#pragma unroll
  for (int t = 0; t < 4; ++t) c1[t] = v8f_zero();

  for (int k0 = 0; k0 < KP1; k0 += KT) {
    __syncthreads();                    // prior tile reads complete
    if (tid < 64) {                     // A tile: 16 rows x 4 x 16B
      int m = tid >> 2, q = tid & 3;
      async_b128(lds_a + m * KTP + q * 8,
                 xb + (size_t)(row0 + m) * KP1 + k0 + q * 8);
    }
#pragma unroll
    for (int j = 0; j < 8; ++j) {       // W1 slice: 512 rows x 4 x 16B
      int i = tid + j * 256;
      int n = i >> 2, q = i & 3;
      async_b128(lds_w + n * KTP + q * 8,
                 w1t + (size_t)n * KP1 + k0 + q * 8);
    }
    wait_async();
    __syncthreads();                    // all waves' DMA visible
    v16bf a = ld_frag(lds_a + m15 * KTP, half);
#pragma unroll
    for (int t = 0; t < 4; ++t) {
      int n0 = wave * 64 + t * 16;
      v16bf b = ld_frag(lds_w + (size_t)(n0 + m15) * KTP, half);
      c1[t] = __builtin_amdgcn_wmma_f32_16x16x32_bf16(false, a, false, b,
                                                      (short)0, c1[t], false, false);
    }
  }
  // write intermediate tile (bias + lrelu) into LDS as bf16
#pragma unroll
  for (int t = 0; t < 4; ++t) {
    int col = wave * 64 + t * 16 + m15;
    float bias = pb1[col];
#pragma unroll
    for (int r = 0; r < 8; ++r) {
      int m = r + 8 * half;
      lds_t[m * PHP + col] = f2bf(lrelu(c1[t][r] + bias));
    }
  }

  // ---- stage B: h = lrelu(t @ W2 + b2), wave w owns cols [32w, 32w+32) ----
  v8f c2[2];
  c2[0] = v8f_zero(); c2[1] = v8f_zero();
  for (int k0 = 0; k0 < PH_; k0 += KT) {
    __syncthreads();
#pragma unroll
    for (int j = 0; j < 4; ++j) {       // W2 slice: 256 rows x 4 x 16B
      int i = tid + j * 256;
      int n = i >> 2, q = i & 3;
      async_b128(lds_w + n * KTP + q * 8,
                 w2t + (size_t)n * PH_ + k0 + q * 8);
    }
    wait_async();
    __syncthreads();
    v16bf a = ld_frag(lds_t + m15 * PHP + k0, half);
#pragma unroll
    for (int t = 0; t < 2; ++t) {
      int n0 = wave * 32 + t * 16;
      v16bf b = ld_frag(lds_w + (size_t)(n0 + m15) * KTP, half);
      c2[t] = __builtin_amdgcn_wmma_f32_16x16x32_bf16(false, a, false, b,
                                                      (short)0, c2[t], false, false);
    }
  }
#pragma unroll
  for (int t = 0; t < 2; ++t) {
    int col = wave * 32 + t * 16 + m15;
    float bias = pb2[col];
#pragma unroll
    for (int r = 0; r < 8; ++r) {
      int m = r + 8 * half;
      float v = lrelu(c2[t][r] + bias);
      h [(size_t)(row0 + m) * 256 + col] = v;
      hb[(size_t)(row0 + m) * 256 + col] = f2bf(v);
    }
  }
}

// ------------------------ generic K=256 WMMA GEMM --------------------------
// out = A[N,256] @ W[256,NCOLS]; A given as bf16 mirror; out bf16 or fp32.
template <int NCOLS, bool OUT_BF16>
__global__ __launch_bounds__(256)
void gemm256_kernel(const unsigned short* __restrict__ Ab,  // [N][256] bf16
                    const unsigned short* __restrict__ Wt,  // [NCOLS][256] bf16
                    float* __restrict__ outF,
                    unsigned short* __restrict__ outB) {
  constexpr int KTOT = 256, KT = 32, KTP = 40;
  constexpr int TILES = NCOLS / 128;              // 16x16 tiles per wave (8 waves)
  __shared__ unsigned short lds_a[16 * KTP];
  __shared__ unsigned short lds_w[NCOLS * KTP];

  const int tid = threadIdx.x, lane = tid & 31, wave = tid >> 5;
  const int m15 = lane & 15, half = lane >> 4;
  const int row0 = blockIdx.x * 16;

  v8f cacc[TILES];
#pragma unroll
  for (int t = 0; t < TILES; ++t) cacc[t] = v8f_zero();

  for (int k0 = 0; k0 < KTOT; k0 += KT) {
    __syncthreads();
    if (tid < 64) {                     // A tile: 16 rows x 4 x 16B
      int m = tid >> 2, q = tid & 3;
      async_b128(lds_a + m * KTP + q * 8,
                 Ab + (size_t)(row0 + m) * KTOT + k0 + q * 8);
    }
#pragma unroll
    for (int j = 0; j < NCOLS / 64; ++j) {   // W slice: NCOLS rows x 4 x 16B
      int i = tid + j * 256;
      int n = i >> 2, q = i & 3;
      async_b128(lds_w + n * KTP + q * 8,
                 Wt + (size_t)n * KTOT + k0 + q * 8);
    }
    wait_async();
    __syncthreads();
    v16bf a = ld_frag(lds_a + m15 * KTP, half);
#pragma unroll
    for (int t = 0; t < TILES; ++t) {
      int n0 = (wave * TILES + t) * 16;
      v16bf b = ld_frag(lds_w + (size_t)(n0 + m15) * KTP, half);
      cacc[t] = __builtin_amdgcn_wmma_f32_16x16x32_bf16(false, a, false, b,
                                                        (short)0, cacc[t], false, false);
    }
  }
#pragma unroll
  for (int t = 0; t < TILES; ++t) {
    int col = (wave * TILES + t) * 16 + m15;
#pragma unroll
    for (int r = 0; r < 8; ++r) {
      int m = r + 8 * half;
      float v = cacc[t][r];
      if constexpr (OUT_BF16) outB[(size_t)(row0 + m) * NCOLS + col] = f2bf(v);
      else                    outF[(size_t)(row0 + m) * NCOLS + col] = v;
    }
  }
}

// ------------------------------ scatter-add --------------------------------
__global__ void zero_kernel(float4* __restrict__ p, int n4) {
  int i = blockIdx.x * blockDim.x + threadIdx.x;
  if (i < n4) p[i] = make_float4(0.f, 0.f, 0.f, 0.f);
}

// One wave per (edge | self-loop); lane owns C/32 contiguous columns.
template <int C>
__global__ void scatter_kernel(const unsigned short* __restrict__ msg,
                               const int* __restrict__ src, const int* __restrict__ dst,
                               const float* __restrict__ dinv, float* __restrict__ acc) {
  int wid = (int)((blockIdx.x * (unsigned)blockDim.x + threadIdx.x) >> 5);
  int lane = threadIdx.x & 31;
  if (wid >= N_EDGES + N_NODES) return;
  int s, d;
  if (wid < N_EDGES) { s = src[wid]; d = dst[wid]; }
  else               { s = wid - N_EDGES; d = s; }
  float w = dinv[s] * dinv[d];
  float* arow = acc + (size_t)d * C;
  int cb = lane * (C / 32);
  if constexpr (C == 256) {               // 8 bf16 per lane: one b128 gather
    uint4 u = reinterpret_cast<const uint4*>(msg + (size_t)s * C)[lane];
    unsigned uu[4] = {u.x, u.y, u.z, u.w};
#pragma unroll
    for (int i = 0; i < 4; ++i) {
      atomicAdd(&arow[cb + 2 * i + 0], __uint_as_float(uu[i] << 16) * w);
      atomicAdd(&arow[cb + 2 * i + 1], __uint_as_float(uu[i] & 0xFFFF0000u) * w);
    }
  } else {                                // 4 bf16 per lane: one b64 gather
    uint2 u = reinterpret_cast<const uint2*>(msg + (size_t)s * C)[lane];
    unsigned uu[2] = {u.x, u.y};
#pragma unroll
    for (int i = 0; i < 2; ++i) {
      atomicAdd(&arow[cb + 2 * i + 0], __uint_as_float(uu[i] << 16) * w);
      atomicAdd(&arow[cb + 2 * i + 1], __uint_as_float(uu[i] & 0xFFFF0000u) * w);
    }
  }
}

// out = lrelu(acc + bias) + resid ; optionally maintain bf16 mirror.
// Processes 4 floats per thread. (conv1/2: out==resid==h, in-place.)
template <bool WRITE_BF16>
__global__ void update_kernel(float* __restrict__ out, const float* __restrict__ resid,
                              const float* __restrict__ acc, const float* __restrict__ bias,
                              unsigned short* __restrict__ hb, int total4, int cmask4) {
  int i = blockIdx.x * blockDim.x + threadIdx.x;
  if (i >= total4) return;
  float4 a = reinterpret_cast<const float4*>(acc)[i];
  float4 r = reinterpret_cast<const float4*>(resid)[i];
  float4 b = reinterpret_cast<const float4*>(bias)[i & cmask4];
  float4 o;
  o.x = lrelu(a.x + b.x) + r.x;
  o.y = lrelu(a.y + b.y) + r.y;
  o.z = lrelu(a.z + b.z) + r.z;
  o.w = lrelu(a.w + b.w) + r.w;
  reinterpret_cast<float4*>(out)[i] = o;
  if constexpr (WRITE_BF16) {
    uint2 p;
    p.x = (unsigned)f2bf(o.x) | ((unsigned)f2bf(o.y) << 16);
    p.y = (unsigned)f2bf(o.z) | ((unsigned)f2bf(o.w) << 16);
    reinterpret_cast<uint2*>(hb)[i] = p;
  }
}

// ------------------------------- max pooling -------------------------------
__global__ void pool_init_kernel(unsigned* __restrict__ pooled) {
  int i = blockIdx.x * blockDim.x + threadIdx.x;
  if (i < N_GRAPH * 128) pooled[i] = 0xFF800000u;     // -inf
}
__global__ void pool_kernel(const float* __restrict__ h3, const int* __restrict__ bid,
                            float* __restrict__ pooled) {
  int i = blockIdx.x * blockDim.x + threadIdx.x;
  if (i >= N_NODES * 128) return;
  int node = i >> 7, c = i & 127;
  float v = h3[i];
  float* a = &pooled[(size_t)bid[node] * 128 + c];
  if (v >= 0.f) atomicMax(reinterpret_cast<int*>(a), __float_as_int(v));
  else          atomicMin(reinterpret_cast<unsigned*>(a), __float_as_uint(v));
}
__global__ void post_kernel(const float* __restrict__ pooled,
                            const float* __restrict__ pw, float* __restrict__ out) {
  __shared__ float red[128];
  int g = blockIdx.x, t = threadIdx.x;
  red[t] = pooled[(size_t)g * 128 + t] * pw[t];
  __syncthreads();
  for (int s = 64; s > 0; s >>= 1) {
    if (t < s) red[t] += red[t + s];
    __syncthreads();
  }
  if (t == 0) out[g] = red[0];
}

// ------------------------------- orchestration -----------------------------
extern "C" void kernel_launch(void* const* d_in, const int* in_sizes, int n_in,
                              void* d_out, int out_size, void* d_ws, size_t ws_size,
                              hipStream_t stream) {
  (void)in_sizes; (void)n_in; (void)out_size;
  const float* x    = (const float*)d_in[0];
  const int*   ei   = (const int*)d_in[1];
  const int*   bid  = (const int*)d_in[2];
  const float* pw1  = (const float*)d_in[3];
  const float* pb1  = (const float*)d_in[4];
  const float* pw2  = (const float*)d_in[5];
  const float* pb2  = (const float*)d_in[6];
  const float* cw1  = (const float*)d_in[7];
  const float* cb1  = (const float*)d_in[8];
  const float* cw2  = (const float*)d_in[9];
  const float* cb2  = (const float*)d_in[10];
  const float* cw3  = (const float*)d_in[11];
  const float* cb3  = (const float*)d_in[12];
  const float* rw3  = (const float*)d_in[13];
  const float* postw= (const float*)d_in[14];
  const int* esrc = ei;
  const int* edst = ei + N_EDGES;

  // workspace carve-up (float units)
  float* W = (float*)d_ws;
  size_t o = 0;
  float* deg = W + o;                    o += N_NODES;                 // -> dinv in-place
  float* h   = W + o;                    o += (size_t)N_NODES * 256;
  float* acc = W + o;                    o += (size_t)N_NODES * 256;
  unsigned short* xb = (unsigned short*)acc;  // bf16 [N][160]; acc unused until conv1
  unsigned short* hb = (unsigned short*)(W + o); o += (size_t)N_NODES * 128; // bf16 h mirror
  unsigned short* msg = (unsigned short*)(W + o);                      // bf16 messages
  float* h3 = (float*)(W + o);           o += (size_t)N_NODES * 128;   // reused post-scatter3
  float* ident = W + o;                  o += (size_t)N_NODES * 128;
  float* pooled = W + o;                 o += (size_t)N_GRAPH * 128;
  unsigned short* w1t = (unsigned short*)(W + o); o += (size_t)512 * 160 / 2;
  unsigned short* w2t = (unsigned short*)(W + o); o += (size_t)256 * 512 / 2;
  unsigned short* c1t = (unsigned short*)(W + o); o += (size_t)256 * 256 / 2;
  unsigned short* c2t = (unsigned short*)(W + o); o += (size_t)256 * 256 / 2;
  unsigned short* c3t = (unsigned short*)(W + o); o += (size_t)128 * 256 / 2;
  unsigned short* r3t = (unsigned short*)(W + o); o += (size_t)128 * 256 / 2;
  if (ws_size < o * sizeof(float)) return;   // insufficient scratch: do nothing (deterministic)

  const int T = 256;
  auto cdiv = [](int a, int b) { return (a + b - 1) / b; };

  // weights + input -> bf16 (transposed / padded)
  convert_wt_kernel<<<cdiv(512 * 160, T), T, 0, stream>>>(pw1, w1t, 140, 160, 512);
  convert_wt_kernel<<<cdiv(256 * 512, T), T, 0, stream>>>(pw2, w2t, 512, 512, 256);
  convert_wt_kernel<<<cdiv(256 * 256, T), T, 0, stream>>>(cw1, c1t, 256, 256, 256);
  convert_wt_kernel<<<cdiv(256 * 256, T), T, 0, stream>>>(cw2, c2t, 256, 256, 256);
  convert_wt_kernel<<<cdiv(128 * 256, T), T, 0, stream>>>(cw3, c3t, 256, 256, 128);
  convert_wt_kernel<<<cdiv(128 * 256, T), T, 0, stream>>>(rw3, r3t, 256, 256, 128);
  convert_x_kernel<<<cdiv(N_NODES * 160, T), T, 0, stream>>>(x, xb);

  // GCN degree normalization (deg becomes dinv in-place)
  deg_init_kernel<<<cdiv(N_NODES, T), T, 0, stream>>>(deg);
  deg_edge_kernel<<<cdiv(N_EDGES, T), T, 0, stream>>>(edst, deg);
  deg_fin_kernel<<<cdiv(N_NODES, T), T, 0, stream>>>(deg);

  // prenet (writes h fp32 + hb bf16)
  prenet_kernel<<<N_NODES / 16, T, 0, stream>>>(xb, w1t, pb1, w2t, pb2, h, hb);

  const int scatterBlocks = (N_EDGES + N_NODES) / 8;   // 1 wave per edge, 8 waves/block

  // conv1
  gemm256_kernel<256, true><<<N_NODES / 16, T, 0, stream>>>(hb, c1t, nullptr, msg);
  zero_kernel<<<cdiv(N_NODES * 64, T), T, 0, stream>>>((float4*)acc, N_NODES * 64);
  scatter_kernel<256><<<scatterBlocks, T, 0, stream>>>(msg, esrc, edst, deg, acc);
  update_kernel<true><<<cdiv(N_NODES * 64, T), T, 0, stream>>>(
      h, h, acc, cb1, hb, N_NODES * 64, 63);

  // conv2
  gemm256_kernel<256, true><<<N_NODES / 16, T, 0, stream>>>(hb, c2t, nullptr, msg);
  zero_kernel<<<cdiv(N_NODES * 64, T), T, 0, stream>>>((float4*)acc, N_NODES * 64);
  scatter_kernel<256><<<scatterBlocks, T, 0, stream>>>(msg, esrc, edst, deg, acc);
  update_kernel<true><<<cdiv(N_NODES * 64, T), T, 0, stream>>>(
      h, h, acc, cb2, hb, N_NODES * 64, 63);

  // conv3 (+ linear residual projection)
  gemm256_kernel<128, true ><<<N_NODES / 16, T, 0, stream>>>(hb, c3t, nullptr, msg);
  gemm256_kernel<128, false><<<N_NODES / 16, T, 0, stream>>>(hb, r3t, ident, nullptr);
  zero_kernel<<<cdiv(N_NODES * 32, T), T, 0, stream>>>((float4*)acc, N_NODES * 32);
  scatter_kernel<128><<<scatterBlocks, T, 0, stream>>>(msg, esrc, edst, deg, acc);
  update_kernel<false><<<cdiv(N_NODES * 32, T), T, 0, stream>>>(
      h3, ident, acc, cb3, nullptr, N_NODES * 32, 31);

  // global max pool + postnet
  pool_init_kernel<<<cdiv(N_GRAPH * 128, T), T, 0, stream>>>((unsigned*)pooled);
  pool_kernel<<<cdiv(N_NODES * 128, T), T, 0, stream>>>(h3, bid, pooled);
  post_kernel<<<N_GRAPH, 128, 0, stream>>>(pooled, postw, (float*)d_out);
}
